// SemiGNN_74071005987284
// MI455X (gfx1250) — matrix operand order repaired and checked by reading, not compiled, not run on hardware
//
#include <hip/hip_runtime.h>
#include <math.h>

// ---------------- problem constants (match reference) ----------------
#define NN      6144
#define DD      128
#define META    2
#define E1      64
#define E2      32
#define E3      64
#define CC      2
#define BB      1024
#define PP      65536
#define NP      144        // padded GEMM width: 128 emb cols + 1 ones col + 15 zero cols

typedef __attribute__((ext_vector_type(16))) __bf16 v16bf;
typedef __attribute__((ext_vector_type(8)))  float  v8f;
typedef __attribute__((ext_vector_type(4)))  float  v4f;

// ---------------- stage 1: scores -> e_j = exp(score - viewmax) ----------------
__global__ __launch_bounds__(1024) void k_scores(const float* __restrict__ emb,
                                                 const float* __restrict__ Hv,
                                                 float* __restrict__ e_out) {
    __shared__ float hv[DD];
    __shared__ float red[1024];
    const int m = blockIdx.x, tid = threadIdx.x;
    if (tid < DD) hv[tid] = Hv[m * DD + tid];
    __syncthreads();
    float sc[NN / 1024];
    float mx = -1e30f;
    for (int q = 0; q < NN / 1024; ++q) {
        const int n = tid + q * 1024;
        const float* er = emb + (size_t)n * DD;
        float s = 0.f;
        #pragma unroll 16
        for (int d = 0; d < DD; ++d) s += er[d] * hv[d];
        sc[q] = s; mx = fmaxf(mx, s);
    }
    red[tid] = mx; __syncthreads();
    for (int off = 512; off > 0; off >>= 1) {
        if (tid < off) red[tid] = fmaxf(red[tid], red[tid + off]);
        __syncthreads();
    }
    const float gmax = red[0];
    for (int q = 0; q < NN / 1024; ++q)
        e_out[m * NN + tid + q * 1024] = expf(sc[q] - gmax);
}

// ---------------- stage 2: build transposed, padded B matrix in bf16 ----------------
// embxT[d][n]: d<128 -> emb[n][d]; d==128 -> 1.0 (row-sum column); d>128 -> 0
__global__ __launch_bounds__(256) void k_embxT(const float* __restrict__ emb,
                                               __bf16* __restrict__ embxT) {
    const int idx = blockIdx.x * 256 + threadIdx.x;
    if (idx >= NP * NN) return;
    const int d = idx / NN, n = idx % NN;
    float v = (d < DD) ? emb[(size_t)n * DD + d] : (d == DD ? 1.0f : 0.0f);
    embxT[idx] = (__bf16)v;
}

// ---------------- stage 3: fused attention GEMM, K-split for MLP ----------------
// grid = (NN/64, S, META); block = 128 threads (4 waves); wave owns a 16-row M tile,
// 9 accumulator tiles (144 cols incl. row-sum col). Raw f32 partials -> accp[kc].
// adj is streamed once with NT hint (protects L2-resident embxT); speculative
// prefetch runs 8 K-steps ahead and covers both 64B lines of each row segment.
__global__ __launch_bounds__(128) void k_attn_gemm(const float* __restrict__ adj,
                                                   const float* __restrict__ e,
                                                   const __bf16* __restrict__ embxT,
                                                   float* __restrict__ accp,
                                                   int kchunk) {
    __shared__ float e_lds[NN];                 // 24 KB of 320 KB LDS
    const int m    = blockIdx.z;
    const int kc   = blockIdx.y;
    const int wave = threadIdx.x >> 5;
    const int lane = threadIdx.x & 31;
    const int half = lane >> 4;
    const int l15  = lane & 15;

    for (int i = threadIdx.x; i < NN; i += 128) e_lds[i] = e[m * NN + i];
    __syncthreads();

    const int    i0       = blockIdx.x * 64 + wave * 16;
    const size_t arow_off = ((size_t)m * NN + (i0 + l15)) * NN;
    const int    k_begin  = kc * kchunk;
    const int    k_end    = k_begin + kchunk;

    v8f acc[9] = {};                            // 72 VGPRs of f32 accumulators

    for (int k0 = k_begin; k0 < k_end; k0 += 32) {
        // ---- A fragment: E~[i, k] = adj * e_k, bf16, ISA 16x32 layout (K interleaved by 8) ----
        v16bf afrag;
        {
            const float* ap = adj + arow_off + k0 + half * 8;   // lanes L & L+16 cover 128B/row
            // speculative stream prefetch, 8 iters ahead; half*16 covers both 64B lines
            __builtin_prefetch(ap + 256 + half * 16, 0, 0);
            const v4f a0 = __builtin_nontemporal_load((const v4f*)ap);
            const v4f a1 = __builtin_nontemporal_load((const v4f*)(ap + 4));
            const v4f a2 = __builtin_nontemporal_load((const v4f*)(ap + 16));
            const v4f a3 = __builtin_nontemporal_load((const v4f*)(ap + 20));
            const float* ep = e_lds + k0 + half * 8;
            #pragma unroll
            for (int j = 0; j < 4; ++j) {
                afrag[j]      = (__bf16)(a0[j] * ep[j]);
                afrag[4 + j]  = (__bf16)(a1[j] * ep[4 + j]);
                afrag[8 + j]  = (__bf16)(a2[j] * ep[16 + j]);
                afrag[12 + j] = (__bf16)(a3[j] * ep[20 + j]);
            }
        }
        // ---- 9 B fragments + WMMA (B: K interleaved by 8, matching A) ----
        #pragma unroll
        for (int t = 0; t < 9; ++t) {
            const __bf16* bp = embxT + (size_t)(t * 16 + l15) * NN + k0 + half * 8;
            v16bf bfrag;
            #pragma unroll
            for (int j = 0; j < 16; ++j) bfrag[j] = bp[j < 8 ? j : j + 8];
            acc[t] = __builtin_amdgcn_wmma_f32_16x16x32_bf16(
                false, afrag, false, bfrag, (short)0, acc[t], false, false);
        }
    }

    // ---- epilogue: store raw partial tiles (deterministic; no FP atomics) ----
    float* obase = accp + ((size_t)kc * META + m) * NN * NP;
    #pragma unroll
    for (int r = 0; r < 8; ++r) {
        const int row = i0 + r + half * 8;
        float* orow = obase + (size_t)row * NP;
        #pragma unroll
        for (int t = 0; t < 9; ++t)
            orow[t * 16 + l15] = acc[t][r];
    }
}

// ---------------- stage 3b: combine K-chunks, divide by row-sum ----------------
__global__ __launch_bounds__(128) void k_combine(const float* __restrict__ accp,
                                                 float* __restrict__ h1, int S) {
    const int i = blockIdx.x, m = blockIdx.y, d = threadIdx.x;
    __shared__ float den;
    const size_t rowoff = ((size_t)m * NN + i) * NP;
    const size_t cstride = (size_t)META * NN * NP;
    float num = 0.f;
    for (int s = 0; s < S; ++s) num += accp[(size_t)s * cstride + rowoff + d];
    if (d == 0) {
        float ds = 0.f;
        for (int s = 0; s < S; ++s) ds += accp[(size_t)s * cstride + rowoff + DD];
        den = ds;
    }
    __syncthreads();
    h1[((size_t)m * NN + i) * DD + d] = num / den;
}

// ---------------- stage 4: per-node MLP + view weighting + projection -> a_u ----------------
__global__ __launch_bounds__(64) void k_mlp(const float* __restrict__ h1,
                                            const float* __restrict__ W1, const float* __restrict__ b1,
                                            const float* __restrict__ W2, const float* __restrict__ b2,
                                            const float* __restrict__ phi,
                                            const float* __restrict__ Wa, const float* __restrict__ ba,
                                            float* __restrict__ a_u) {
    __shared__ float row[META][DD];
    __shared__ float ha[META][E1];
    __shared__ float h2[META * E2];
    const int i = blockIdx.x, t = threadIdx.x;

    #pragma unroll
    for (int m = 0; m < META; ++m) {
        row[m][t]      = h1[((size_t)m * NN + i) * DD + t];
        row[m][t + 64] = h1[((size_t)m * NN + i) * DD + t + 64];
    }
    __syncthreads();
    #pragma unroll
    for (int m = 0; m < META; ++m) {
        float s = b1[m * E1 + t];
        const float* w = W1 + (size_t)m * DD * E1;
        for (int d = 0; d < DD; ++d) s += row[m][d] * w[d * E1 + t];
        ha[m][t] = fmaxf(s, 0.f);
    }
    __syncthreads();
    // vw = softmax(phi)
    const float p0 = phi[0], p1 = phi[1];
    const float pm = fmaxf(p0, p1);
    const float ex0 = expf(p0 - pm), ex1 = expf(p1 - pm);
    const float vw0 = ex0 / (ex0 + ex1), vw1 = ex1 / (ex0 + ex1);
    {
        const int m = t >> 5, f = t & 31;
        float s = b2[m * E2 + f];
        const float* w = W2 + (size_t)m * E1 * E2;
        for (int d = 0; d < E1; ++d) s += ha[m][d] * w[d * E2 + f];
        s = fmaxf(s, 0.f);
        h2[m * E2 + f] = s * (m ? vw1 : vw0);
    }
    __syncthreads();
    {
        float s = ba[t];
        for (int f = 0; f < META * E2; ++f) s += h2[f] * Wa[f * E3 + t];
        a_u[(size_t)i * E3 + t] = s;
    }
}

// ---------------- stage 5: loss1 + accuracy (B = 1024, one workgroup) ----------------
__global__ __launch_bounds__(1024) void k_loss1(const float* __restrict__ a_u,
                                                const int* __restrict__ idx_mask,
                                                const float* __restrict__ label,
                                                const float* __restrict__ u_param,
                                                float* __restrict__ loss1_ws,
                                                float* __restrict__ out) {
    __shared__ float r1[1024];
    __shared__ float r2[1024];
    const int b = threadIdx.x;
    const int idx = idx_mask[b];
    const float* md = a_u + (size_t)idx * E3;
    float l0 = 0.f, l1 = 0.f;
    for (int d = 0; d < E3; ++d) { const float v = md[d]; l0 += v * u_param[d * CC]; l1 += v * u_param[d * CC + 1]; }
    const float mxl = fmaxf(l0, l1);
    const float ee0 = expf(l0 - mxl), ee1 = expf(l1 - mxl);
    const float p0 = ee0 / (ee0 + ee1), p1 = ee1 / (ee0 + ee1);
    // faithful double softmax: q = log_softmax(p)
    const float mp = fmaxf(p0, p1);
    const float lse = mp + logf(expf(p0 - mp) + expf(p1 - mp));
    const float q0 = p0 - lse, q1 = p1 - lse;
    const float ml0 = label[idx * CC], ml1 = label[idx * CC + 1];
    r1[b] = ml0 * q0 + ml1 * q1;
    const int pred   = (p1 > p0) ? 1 : 0;     // argmax, first index on tie
    const int actual = (ml1 > ml0) ? 1 : 0;
    r2[b] = (pred == actual) ? 1.f : 0.f;
    __syncthreads();
    for (int off = 512; off > 0; off >>= 1) {
        if (b < off) { r1[b] += r1[b + off]; r2[b] += r2[b + off]; }
        __syncthreads();
    }
    if (b == 0) { loss1_ws[0] = -r1[0] / (float)BB; out[1] = r2[0] / (float)BB; }
}

// ---------------- stage 6: loss2 partials over P pairs ----------------
__global__ __launch_bounds__(256) void k_loss2(const float* __restrict__ a_u,
                                               const int* __restrict__ u_i,
                                               const int* __restrict__ u_j,
                                               const float* __restrict__ gl,
                                               float* __restrict__ part) {
    __shared__ float red[256];
    const int p = blockIdx.x * 256 + threadIdx.x;
    const float* x = a_u + (size_t)u_i[p] * E3;
    const float* y = a_u + (size_t)u_j[p] * E3;
    float s = 0.f;
    #pragma unroll 8
    for (int d = 0; d < E3; ++d) s += x[d] * y[d];
    const float z = gl[p] * s;
    red[threadIdx.x] = fminf(z, 0.f) - log1pf(expf(-fabsf(z)));   // log_sigmoid, stable
    __syncthreads();
    for (int off = 128; off > 0; off >>= 1) {
        if (threadIdx.x < off) red[threadIdx.x] += red[threadIdx.x + off];
        __syncthreads();
    }
    if (threadIdx.x == 0) part[blockIdx.x] = red[0];
}

// ---------------- stage 7: combine losses ----------------
__global__ __launch_bounds__(256) void k_final(const float* __restrict__ part,
                                               const float* __restrict__ loss1_ws,
                                               float* __restrict__ out) {
    __shared__ float red[256];
    red[threadIdx.x] = part[threadIdx.x];
    __syncthreads();
    for (int off = 128; off > 0; off >>= 1) {
        if (threadIdx.x < off) red[threadIdx.x] += red[threadIdx.x + off];
        __syncthreads();
    }
    if (threadIdx.x == 0) {
        const float loss2 = -red[0] / (float)PP;
        out[0] = 0.5f * loss1_ws[0] + 0.5f * loss2;
    }
}

// ---------------- host launcher ----------------
extern "C" void kernel_launch(void* const* d_in, const int* in_sizes, int n_in,
                              void* d_out, int out_size, void* d_ws, size_t ws_size,
                              hipStream_t stream) {
    const float* adj      = (const float*)d_in[0];
    const float* emb      = (const float*)d_in[1];
    const float* H_v      = (const float*)d_in[2];
    const float* W1       = (const float*)d_in[3];
    const float* b1       = (const float*)d_in[4];
    const float* W2       = (const float*)d_in[5];
    const float* b2       = (const float*)d_in[6];
    const float* phi      = (const float*)d_in[7];
    const float* Wa       = (const float*)d_in[8];
    const float* ba       = (const float*)d_in[9];
    const float* u_param  = (const float*)d_in[10];
    const float* glabel   = (const float*)d_in[11];
    const float* label    = (const float*)d_in[12];
    const int*   u_i      = (const int*)d_in[13];
    const int*   u_j      = (const int*)d_in[14];
    const int*   idx_mask = (const int*)d_in[15];
    float* out = (float*)d_out;

    // workspace layout (all offsets 256B aligned)
    char* ws = (char*)d_ws;
    const size_t off_e      = 0;                  // 2*6144 f32         = 48 KB
    const size_t off_embxT  = 49152;              // 144*6144 bf16      = 1.69 MB
    const size_t off_au     = 1818624;            // 6144*64 f32        = 1.5 MB
    const size_t off_part   = 3391488;            // 256 f32
    const size_t off_loss1  = 3392512;            // 1 f32 (+pad)
    const size_t off_h1     = 3392768;            // 2*6144*128 f32     = 6.0 MB
    const size_t off_accp   = 9684224;            // S * 2*6144*144 f32 = S * 6.75 MB
    const size_t accp_bytes = (size_t)META * NN * NP * sizeof(float);   // per K-chunk

    // K-split factor: more chunks -> more waves in flight on the adj HBM stream.
    int S = 1;
    if (ws_size >= off_accp + 4 * accp_bytes)      S = 4;
    else if (ws_size >= off_accp + 2 * accp_bytes) S = 2;

    float*  e_buf   = (float*)(ws + off_e);
    __bf16* embxT   = (__bf16*)(ws + off_embxT);
    float*  au_buf  = (float*)(ws + off_au);
    float*  part    = (float*)(ws + off_part);
    float*  loss1ws = (float*)(ws + off_loss1);
    float*  h1_buf  = (float*)(ws + off_h1);
    float*  accp    = (float*)(ws + off_accp);

    k_scores<<<META, 1024, 0, stream>>>(emb, H_v, e_buf);
    k_embxT<<<(NP * NN + 255) / 256, 256, 0, stream>>>(emb, embxT);
    {
        dim3 grid(NN / 64, S, META);
        k_attn_gemm<<<grid, 128, 0, stream>>>(adj, e_buf, embxT, accp, NN / S);
    }
    {
        dim3 grid(NN, META);
        k_combine<<<grid, DD, 0, stream>>>(accp, h1_buf, S);
    }
    k_mlp<<<NN, 64, 0, stream>>>(h1_buf, W1, b1, W2, b2, phi, Wa, ba, au_buf);
    k_loss1<<<1, 1024, 0, stream>>>(au_buf, idx_mask, label, u_param, loss1ws, out);
    k_loss2<<<PP / 256, 256, 0, stream>>>(au_buf, u_i, u_j, glabel, part);
    k_final<<<1, 256, 0, stream>>>(part, loss1ws, out);
}
